// StackedMamba_20031727468552
// MI455X (gfx1250) — matrix operand order repaired and checked by reading, not compiled, not run on hardware
//
#include <hip/hip_runtime.h>
#include <hip/hip_bf16.h>
#include <math.h>

#define NB        2
#define D_MODEL   512
#define D_STATE   128
#define D_CONVW   4
#define HEADDIM   64
#define D_INNER   1024
#define NHEADS    16
#define CONV_DIM  1280
#define D_IN_PROJ 2320
#define LABEL     1000
#define BATCH     2
#define SEQ       512
#define ROWS      (BATCH * SEQ)   // 1024
#define TN        64              // N-tile per block in the GEMM

typedef __bf16 bf16_t;
typedef __attribute__((ext_vector_type(16))) __bf16 v16bf;
typedef __attribute__((ext_vector_type(8)))  __bf16 v8bf;
typedef __attribute__((ext_vector_type(8)))  float  v8f;

// ---------------------------------------------------------------- helpers
__device__ __forceinline__ bf16_t f2bf(float f) {
  unsigned u = __float_as_uint(f);
  unsigned short s = (unsigned short)((u + 0x7FFFu + ((u >> 16) & 1u)) >> 16);
  bf16_t b;
  __builtin_memcpy(&b, &s, sizeof(s));
  return b;
}

__device__ __forceinline__ float sigmoidf_(float x) { return 1.f / (1.f + expf(-x)); }

__device__ __forceinline__ v16bf cat8(v8bf lo, v8bf hi) {
  return __builtin_shufflevector(lo, hi, 0,1,2,3,4,5,6,7,8,9,10,11,12,13,14,15);
}

// ---------------------------------------------------------------- kernels
__global__ void embed_kernel(const int* __restrict__ tokens,
                             const float* __restrict__ emb,
                             float* __restrict__ xf, bf16_t* __restrict__ xb) {
  int idx = blockIdx.x * blockDim.x + threadIdx.x;      // ROWS*D_MODEL
  int row = idx / D_MODEL, d = idx % D_MODEL;
  float v = emb[(size_t)tokens[row] * D_MODEL + d];
  xf[idx] = v;
  xb[idx] = f2bf(v);
}

__global__ void cast_kernel(const float* __restrict__ in, bf16_t* __restrict__ out, int n) {
  int idx = blockIdx.x * blockDim.x + threadIdx.x;
  if (idx < n) out[idx] = f2bf(in[idx]);
}

// C[M,N] = A[M,K](bf16) @ W[N,K]^T(bf16) + bias
// Block = 256 threads (8 waves) computing a 128M x 64N tile.
// Per k-step the 64x32 B-slice is DMA'd straight into LDS with
// GLOBAL_LOAD_ASYNC_TO_LDS_B128 (ASYNCcnt), then each wave preloads 4 B
// fragments (batched ds_load_b128 clause) and issues 4 WMMAs per A fragment.
__global__ void gemm_bf16_wmma(const bf16_t* __restrict__ A, const bf16_t* __restrict__ W,
                               const float* __restrict__ bias, float* __restrict__ C,
                               int M, int N, int K) {
  __shared__ __align__(16) bf16_t Bs[TN * 32];   // [n_local][k_local]; only LDS object => offset 0
  const int tid  = threadIdx.x;
  const int wave = tid >> 5, lane = tid & 31;
  const int m0 = (blockIdx.y * 8 + wave) * 16;
  const int nb = blockIdx.x * TN;

  // cooperative B staging: thread -> (row srow, 8 bf16 chunk skc); 16B per lane
  const int srow = tid >> 2;
  const int skc  = (tid & 3) << 3;
  int gn = nb + srow; if (gn >= N) gn = N - 1;   // clamp; OOB cols masked at store
  const bf16_t* gB = W + (size_t)gn * K + skc;
  const unsigned ldsOff = (unsigned)((srow * 32 + skc) * sizeof(bf16_t)); // 16B aligned

  // this lane's A pointer (ISA 7.12.2 A layout)
  const int arow = m0 + (lane & 15);
  const int akb  = (lane >> 4) << 3;             // 0 or 8
  const bf16_t* gA = A + (size_t)arow * K + akb;

  // this lane's B-fragment position inside the LDS tile
  const int bn  = lane & 15;
  const int bkb = (lane >> 4) << 3;              // 0 or 8

  v8f acc[4] = {v8f{}, v8f{}, v8f{}, v8f{}};

  for (int k0 = 0; k0 < K; k0 += 32) {
    // async DMA: global -> LDS, no VGPR round trip (tracked by ASYNCcnt).
    // A pointer to Bs is passed into the asm (unreferenced in the template) so
    // the __shared__ object escapes and the compiler must assume the asm
    // writes it; otherwise the later ds_loads get folded away. A direct "m"
    // operand on LDS crashes the AMDGPU isel, so escape-by-register it is.
    asm volatile("global_load_async_to_lds_b128 %0, %1, off"
                 :
                 : "v"(ldsOff),
                   "v"((unsigned long long)(size_t)(gB + k0)),
                   "v"(Bs)
                 : "memory");
    asm volatile("s_wait_asynccnt 0x0" ::: "memory");
    __syncthreads();

    v8bf alo = *(const v8bf*)(gA + k0);          // K akb..akb+7
    v8bf ahi = *(const v8bf*)(gA + k0 + 16);     // K akb+16..akb+23
    v16bf a = cat8(alo, ahi);

    if (k0 + 32 < K) {                           // stream next slices toward L2/L0
      __builtin_prefetch(gA + k0 + 32, 0, 1);
      __builtin_prefetch(gB + k0 + 32, 0, 1);
    }

    // preload all four B fragments so the ds_load_b128s clause together
    const bf16_t* lb0 = &Bs[(0 * 16 + bn) * 32 + bkb];
    const bf16_t* lb1 = &Bs[(1 * 16 + bn) * 32 + bkb];
    const bf16_t* lb2 = &Bs[(2 * 16 + bn) * 32 + bkb];
    const bf16_t* lb3 = &Bs[(3 * 16 + bn) * 32 + bkb];
    v16bf b0 = cat8(*(const v8bf*)lb0, *(const v8bf*)(lb0 + 16));
    v16bf b1 = cat8(*(const v8bf*)lb1, *(const v8bf*)(lb1 + 16));
    v16bf b2 = cat8(*(const v8bf*)lb2, *(const v8bf*)(lb2 + 16));
    v16bf b3 = cat8(*(const v8bf*)lb3, *(const v8bf*)(lb3 + 16));

    acc[0] = __builtin_amdgcn_wmma_f32_16x16x32_bf16(false, a, false, b0, (short)0, acc[0], false, false);
    acc[1] = __builtin_amdgcn_wmma_f32_16x16x32_bf16(false, a, false, b1, (short)0, acc[1], false, false);
    acc[2] = __builtin_amdgcn_wmma_f32_16x16x32_bf16(false, a, false, b2, (short)0, acc[2], false, false);
    acc[3] = __builtin_amdgcn_wmma_f32_16x16x32_bf16(false, a, false, b3, (short)0, acc[3], false, false);
    __syncthreads();
  }

  const int rbase = m0 + ((lane >> 4) << 3);
#pragma unroll
  for (int j = 0; j < 4; ++j) {
    int col = nb + j * 16 + (lane & 15);
    if (col < N) {
      float bv = bias ? bias[col] : 0.f;
#pragma unroll
      for (int v = 0; v < 8; ++v)
        C[(size_t)(rbase + v) * N + col] = acc[j][v] + bv;
    }
  }
}

// dt = softplus(raw + dt_bias); dA = exp(dt * (-exp(A_log)))
__global__ void dt_kernel(const float* __restrict__ zx, const float* __restrict__ dtb,
                          const float* __restrict__ Alog,
                          float* __restrict__ dt_o, float* __restrict__ dA_o) {
  int idx = blockIdx.x * blockDim.x + threadIdx.x;      // ROWS*NHEADS
  int row = idx / NHEADS, h = idx % NHEADS;
  float raw = zx[(size_t)row * D_IN_PROJ + D_INNER + CONV_DIM + h] + dtb[h];
  float dt = (raw > 20.f) ? raw : log1pf(expf(raw));
  float A  = -expf(Alog[h]);
  dt_o[idx] = dt;
  dA_o[idx] = expf(dt * A);
}

// depthwise causal conv (width 4) + bias + SiLU over the xBC slice of zxbcdt
__global__ void conv_kernel(const float* __restrict__ zx, const float* __restrict__ cw,
                            const float* __restrict__ cb, float* __restrict__ out) {
  int idx = blockIdx.x * blockDim.x + threadIdx.x;      // ROWS*CONV_DIM
  int c = idx % CONV_DIM;
  int l = (idx / CONV_DIM) % SEQ;
  int b = idx / (CONV_DIM * SEQ);
  float acc = cb[c];
#pragma unroll
  for (int k = 0; k < D_CONVW; ++k) {
    int t = l - (D_CONVW - 1) + k;
    if (t >= 0)
      acc += zx[((size_t)(b * SEQ + t)) * D_IN_PROJ + D_INNER + c] * cw[c * D_CONVW + k];
  }
  out[idx] = acc * sigmoidf_(acc);
}

// sequential selective scan: one block per (head, batch); thread owns h[p, n0..n0+31]
__global__ void scan_kernel(const float* __restrict__ xbc, const float* __restrict__ dtb,
                            const float* __restrict__ dab, float* __restrict__ y) {
  const int h = blockIdx.x, b = blockIdx.y;
  const int tid = threadIdx.x;         // 256
  const int p = tid >> 2, q = tid & 3, n0 = q * 32;
  __shared__ float Bs[D_STATE], Cs[D_STATE], xs[HEADDIM];

  float hreg[32];
#pragma unroll
  for (int j = 0; j < 32; ++j) hreg[j] = 0.f;

  for (int t = 0; t < SEQ; ++t) {
    const int row = b * SEQ + t;
    const float dt_t = dtb[row * NHEADS + h];
    const float dA_t = dab[row * NHEADS + h];
    const float* xr = xbc + (size_t)row * CONV_DIM;
    if (tid < D_STATE) Bs[tid] = xr[D_INNER + tid];
    else               Cs[tid - D_STATE] = xr[D_INNER + D_STATE + (tid - D_STATE)];
    if (tid < HEADDIM) xs[tid] = dt_t * xr[h * HEADDIM + tid];
    __syncthreads();
    const float dtx = xs[p];
    float part = 0.f;
#pragma unroll
    for (int j = 0; j < 32; ++j) {
      hreg[j] = fmaf(hreg[j], dA_t, dtx * Bs[n0 + j]);
      part = fmaf(hreg[j], Cs[n0 + j], part);
    }
    part += __shfl_xor(part, 1, 32);   // quad reduction across n-chunks (wave32)
    part += __shfl_xor(part, 2, 32);
    if (q == 0) y[(size_t)row * D_INNER + h * HEADDIM + p] = part;
    __syncthreads();
  }
}

// y = (y_scan + D*xh) * silu(z); RMSNorm over D_INNER; emit bf16
__global__ void gate_rms_kernel(const float* __restrict__ y, const float* __restrict__ xbc,
                                const float* __restrict__ zx, const float* __restrict__ Dv,
                                const float* __restrict__ rmsw, bf16_t* __restrict__ out) {
  int row = blockIdx.x, tid = threadIdx.x;
  __shared__ float red[8];
  const float* yr = y + (size_t)row * D_INNER;
  const float* xr = xbc + (size_t)row * CONV_DIM;
  const float* zr = zx + (size_t)row * D_IN_PROJ;
  float g[4]; float ss = 0.f;
#pragma unroll
  for (int j = 0; j < 4; ++j) {
    int d = tid + j * 256;
    float v = yr[d] + Dv[d >> 6] * xr[d];
    float z = zr[d];
    v *= z * sigmoidf_(z);
    g[j] = v; ss += v * v;
  }
#pragma unroll
  for (int o = 16; o > 0; o >>= 1) ss += __shfl_xor(ss, o, 32);
  if ((tid & 31) == 0) red[tid >> 5] = ss;
  __syncthreads();
  float tot = 0.f;
#pragma unroll
  for (int w = 0; w < 8; ++w) tot += red[w];
  float scale = rsqrtf(tot / (float)D_INNER + 1e-5f);
#pragma unroll
  for (int j = 0; j < 4; ++j) {
    int d = tid + j * 256;
    out[(size_t)row * D_INNER + d] = f2bf(g[j] * scale * rmsw[d]);
  }
}

// x = layernorm(yproj + x); emit fp32 (residual) and bf16 (next GEMM)
__global__ void residual_ln_kernel(const float* __restrict__ yproj,
                                   const float* __restrict__ lnw, const float* __restrict__ lnb,
                                   float* __restrict__ xf, bf16_t* __restrict__ xb) {
  int row = blockIdx.x, tid = threadIdx.x;
  __shared__ float red[8];
  int d0 = tid, d1 = tid + 256;
  size_t base = (size_t)row * D_MODEL;
  float v0 = yproj[base + d0] + xf[base + d0];
  float v1 = yproj[base + d1] + xf[base + d1];
  float s = v0 + v1;
#pragma unroll
  for (int o = 16; o > 0; o >>= 1) s += __shfl_xor(s, o, 32);
  if ((tid & 31) == 0) red[tid >> 5] = s;
  __syncthreads();
  float tot = 0.f;
#pragma unroll
  for (int w = 0; w < 8; ++w) tot += red[w];
  float mu = tot / (float)D_MODEL;
  float c0 = v0 - mu, c1 = v1 - mu;
  float vs = c0 * c0 + c1 * c1;
  __syncthreads();                    // red[] reuse
#pragma unroll
  for (int o = 16; o > 0; o >>= 1) vs += __shfl_xor(vs, o, 32);
  if ((tid & 31) == 0) red[tid >> 5] = vs;
  __syncthreads();
  float vtot = 0.f;
#pragma unroll
  for (int w = 0; w < 8; ++w) vtot += red[w];
  float inv = rsqrtf(vtot / (float)D_MODEL + 1e-5f);
  float o0 = c0 * inv * lnw[d0] + lnb[d0];
  float o1 = c1 * inv * lnw[d1] + lnb[d1];
  xf[base + d0] = o0; xf[base + d1] = o1;
  xb[base + d0] = f2bf(o0); xb[base + d1] = f2bf(o1);
}

// ---------------------------------------------------------------- launch
extern "C" void kernel_launch(void* const* d_in, const int* in_sizes, int n_in,
                              void* d_out, int out_size, void* d_ws, size_t ws_size,
                              hipStream_t stream) {
  const int*   tokens  = (const int*)  d_in[0];
  const float* emb     = (const float*)d_in[1];
  const float* Wi_all  = (const float*)d_in[2];
  const float* cw_all  = (const float*)d_in[3];
  const float* cb_all  = (const float*)d_in[4];
  const float* dtb_all = (const float*)d_in[5];
  const float* Al_all  = (const float*)d_in[6];
  const float* D_all   = (const float*)d_in[7];
  const float* rw_all  = (const float*)d_in[8];
  const float* Wo_all  = (const float*)d_in[9];
  const float* lnw_all = (const float*)d_in[10];
  const float* lnb_all = (const float*)d_in[11];
  const float* hw      = (const float*)d_in[12];
  const float* hb      = (const float*)d_in[13];

  char* ws = (char*)d_ws;
  size_t off = 0;
  auto carve = [&](size_t bytes) -> void* {
    void* p = ws + off;
    off += (bytes + 255) & ~(size_t)255;
    return p;
  };
  float*  x_f32  = (float*) carve((size_t)ROWS * D_MODEL * 4);
  bf16_t* x_bf   = (bf16_t*)carve((size_t)ROWS * D_MODEL * 2);
  float*  zx     = (float*) carve((size_t)ROWS * D_IN_PROJ * 4);
  float*  xbc    = (float*) carve((size_t)ROWS * CONV_DIM * 4);
  float*  dt_buf = (float*) carve((size_t)ROWS * NHEADS * 4);
  float*  dA_buf = (float*) carve((size_t)ROWS * NHEADS * 4);
  float*  y_buf  = (float*) carve((size_t)ROWS * D_INNER * 4);
  bf16_t* yn_bf  = (bf16_t*)carve((size_t)ROWS * D_INNER * 2);
  float*  yproj  = (float*) carve((size_t)ROWS * D_MODEL * 4);
  bf16_t* w_bf   = (bf16_t*)carve((size_t)D_IN_PROJ * D_MODEL * 2);

  embed_kernel<<<ROWS * D_MODEL / 256, 256, 0, stream>>>(tokens, emb, x_f32, x_bf);

  for (int i = 0; i < NB; ++i) {
    const float* Wi  = Wi_all  + (size_t)i * D_IN_PROJ * D_MODEL;
    const float* cw  = cw_all  + (size_t)i * CONV_DIM * D_CONVW;
    const float* cb  = cb_all  + (size_t)i * CONV_DIM;
    const float* dtb = dtb_all + (size_t)i * NHEADS;
    const float* Al  = Al_all  + (size_t)i * NHEADS;
    const float* Dv  = D_all   + (size_t)i * NHEADS;
    const float* rw  = rw_all  + (size_t)i * D_INNER;
    const float* Wo  = Wo_all  + (size_t)i * D_MODEL * D_INNER;
    const float* lnw = lnw_all + (size_t)i * D_MODEL;
    const float* lnb = lnb_all + (size_t)i * D_MODEL;

    cast_kernel<<<(D_IN_PROJ * D_MODEL + 255) / 256, 256, 0, stream>>>(Wi, w_bf, D_IN_PROJ * D_MODEL);
    gemm_bf16_wmma<<<dim3((D_IN_PROJ + TN - 1) / TN, ROWS / 128), 256, 0, stream>>>(
        x_bf, w_bf, nullptr, zx, ROWS, D_IN_PROJ, D_MODEL);

    dt_kernel<<<ROWS * NHEADS / 256, 256, 0, stream>>>(zx, dtb, Al, dt_buf, dA_buf);
    conv_kernel<<<ROWS * CONV_DIM / 256, 256, 0, stream>>>(zx, cw, cb, xbc);
    scan_kernel<<<dim3(NHEADS, BATCH), 256, 0, stream>>>(xbc, dt_buf, dA_buf, y_buf);
    gate_rms_kernel<<<ROWS, 256, 0, stream>>>(y_buf, xbc, zx, Dv, rw, yn_bf);

    cast_kernel<<<(D_MODEL * D_INNER + 255) / 256, 256, 0, stream>>>(Wo, w_bf, D_MODEL * D_INNER);
    gemm_bf16_wmma<<<dim3((D_MODEL + TN - 1) / TN, ROWS / 128), 256, 0, stream>>>(
        yn_bf, w_bf, nullptr, yproj, ROWS, D_MODEL, D_INNER);

    residual_ln_kernel<<<ROWS, 256, 0, stream>>>(yproj, lnw, lnb, x_f32, x_bf);
  }

  cast_kernel<<<(LABEL * D_MODEL + 255) / 256, 256, 0, stream>>>(hw, w_bf, LABEL * D_MODEL);
  gemm_bf16_wmma<<<dim3((LABEL + TN - 1) / TN, ROWS / 128), 256, 0, stream>>>(
      x_bf, w_bf, hb, (float*)d_out, ROWS, LABEL, D_MODEL);
}
